// GroupedQueryAttention_81922206204114
// MI455X (gfx1250) — matrix operand order repaired and checked by reading, not compiled, not run on hardware
//
#include <hip/hip_runtime.h>
#include <stdint.h>

typedef __attribute__((ext_vector_type(16))) _Float16 v16h;
typedef __attribute__((ext_vector_type(8)))  float    v8f;
typedef __attribute__((ext_vector_type(4)))  unsigned int u32x4;
typedef __attribute__((ext_vector_type(8)))  int      i32x8;
typedef __attribute__((ext_vector_type(4)))  int      i32x4;

#define D_MODEL  2048
#define N_HEADS  16
#define N_KV     4
#define HEAD_DIM 128
#define BATCH    2
#define SEQ      2048
#define NTOK     (BATCH * SEQ)   // 4096

__device__ __forceinline__ v8f wmma_f16(v16h a, v16h b, v8f c) {
  return __builtin_amdgcn_wmma_f32_16x16x32_f16(false, a, false, b, (short)0, c,
                                                false, false);
}

// ---------------------------------------------------------------------------
// Tensor Data Mover: 2D tile (tile_w x tile_h elements, 2-byte data) from
// global (row stride row_stride elements) into LDS at lds_off, packed rows.
// D# layout per CDNA5 ISA. Tracked by TENSORcnt.
// ---------------------------------------------------------------------------
#if __has_include(<hip/amd_detail/amd_gfx1250_TDM.h>)
#define TDM_CALL(g0, g1, g2, g3) \
  __builtin_amdgcn_tensor_load_to_lds(g0, g1, g2, g3, (i32x8){0,0,0,0,0,0,0,0}, 0)
#else
#define TDM_CALL(g0, g1, g2, g3) \
  __builtin_amdgcn_tensor_load_to_lds(g0, g1, g2, g3, 0)
#endif

__device__ __forceinline__ void tdm_load_2d(uint32_t lds_off, const void* gptr,
                                            uint32_t tensor_w, uint32_t tensor_h,
                                            uint32_t tile_w, uint32_t tile_h,
                                            uint32_t row_stride) {
  unsigned long long ga = (unsigned long long)gptr;
  u32x4 g0;
  g0[0] = 1u;                                   // count=1 (valid descriptor)
  g0[1] = lds_off;                              // lds_addr (bytes)
  g0[2] = (uint32_t)ga;                         // global_addr[31:0]
  g0[3] = (uint32_t)((ga >> 32) & 0x1FFFFFFu)   // global_addr[56:32]
          | 0x80000000u;                        // type=2 ("image")
  i32x8 g1;
  g1[0] = (int)(1u << 16);                      // data_size=1 -> 2 bytes
  g1[1] = (int)((tensor_w & 0xFFFFu) << 16);    // tensor_dim0[15:0] @bit48
  g1[2] = (int)((tensor_w >> 16) | ((tensor_h & 0xFFFFu) << 16)); // dim0 hi, dim1 lo
  g1[3] = (int)((tensor_h >> 16) | (tile_w << 16));               // dim1 hi, tile_dim0
  g1[4] = (int)tile_h;                          // tile_dim1 (tile_dim2=0)
  g1[5] = (int)row_stride;                      // tensor_dim0_stride[31:0]
  g1[6] = 0;                                    // stride hi / dim1_stride lo
  g1[7] = 0;
  i32x4 z = {0, 0, 0, 0};
  TDM_CALL(g0, g1, z, z);
}

__device__ __forceinline__ uint32_t lds_offset_of(const void* p) {
  return (uint32_t)(uintptr_t)p;                // low 32 bits = LDS byte offset
}

// ---------------------------------------------------------------------------
// fp32 -> f16 cast (one-time reformat of x and weights; memory-bound)
// ---------------------------------------------------------------------------
__global__ void cast_h(const float* __restrict__ src, _Float16* __restrict__ dst,
                       int n) {
  int i = blockIdx.x * blockDim.x + threadIdx.x;
  if (i < n) dst[i] = (_Float16)src[i];
}

// ---------------------------------------------------------------------------
// C[MxN] = A[MxK] * B[KxN], all f16 operands, fp32 accumulate, OutT store.
// One wave per 16x64 strip: each A fetch feeds 4 WMMAs (4 accumulators).
// B slabs (32x64) staged via TDM, double buffered.
// ---------------------------------------------------------------------------
template <typename OutT>
__global__ void __launch_bounds__(32)
gemm_h16(const _Float16* __restrict__ A, const _Float16* __restrict__ B,
         OutT* __restrict__ C, int M, int N, int K) {
  __shared__ __align__(128) _Float16 smemB[2][32 * 64];
  const int lane  = threadIdx.x & 31;
  const int half_ = lane >> 4;
  const int lm    = lane & 15;
  const int n0    = blockIdx.x * 64;
  const int m0    = blockIdx.y * 16;
  const _Float16* Arow = A + (size_t)(m0 + lm) * K;
  const uint32_t lds0 = lds_offset_of(&smemB[0][0]);
  const uint32_t lds1 = lds_offset_of(&smemB[1][0]);

  // prologue: fetch B slab 0 (32 rows x 64 cols)
  tdm_load_2d(lds0, B + n0, (uint32_t)N, 32u, 64u, 32u, (uint32_t)N);

  v8f acc[4];
  v8f zacc = {};
#pragma unroll
  for (int j = 0; j < 4; ++j) acc[j] = zacc;

  const int nk = K / 32;
  for (int t = 0; t < nk; ++t) {
    if (t + 1 < nk) {
      tdm_load_2d((t & 1) ? lds0 : lds1,
                  B + (size_t)(t + 1) * 32 * N + n0,
                  (uint32_t)N, 32u, 64u, 32u, (uint32_t)N);
      __builtin_amdgcn_s_wait_tensorcnt(1);   // slab t ready, t+1 in flight
    } else {
      __builtin_amdgcn_s_wait_tensorcnt(0);
    }
    const int k0 = t * 32;
    v16h a;
#pragma unroll
    for (int e = 0; e < 8; ++e) {             // two contiguous b128 loads
      a[e]     = Arow[k0 + half_ * 16 + e];
      a[e + 8] = Arow[k0 + half_ * 16 + 8 + e];
    }
    const _Float16* bb = &smemB[t & 1][0];
#pragma unroll
    for (int j = 0; j < 4; ++j) {
      v16h b;
#pragma unroll
      for (int e = 0; e < 16; ++e)            // K on elems, N on lanes
        b[e] = bb[(half_ * 16 + e) * 64 + j * 16 + lm];
      acc[j] = wmma_f16(a, b, acc[j]);
    }
  }
#pragma unroll
  for (int j = 0; j < 4; ++j)
#pragma unroll
    for (int r = 0; r < 8; ++r)
      C[(size_t)(m0 + r + half_ * 8) * N + (n0 + j * 16 + lm)] = (OutT)acc[j][r];
}

// ---------------------------------------------------------------------------
// RoPE (base 10000, head_dim 128), f16 in -> f16 out, fp32 math.
// ---------------------------------------------------------------------------
__global__ void rope_h(const _Float16* __restrict__ src,
                       _Float16* __restrict__ dst, int nheads, int total) {
  int idx = blockIdx.x * blockDim.x + threadIdx.x;
  if (idx >= total) return;
  int i   = idx & 63;
  int h   = (idx >> 6) % nheads;
  int tok = idx / (64 * nheads);
  int t   = tok % SEQ;
  float ang = (float)t * __expf(-(float)i * 0.14391156f);  // ln(1e4)/64
  float c = __cosf(ang), s = __sinf(ang);
  size_t base = ((size_t)tok * nheads + h) * HEAD_DIM;
  float x1 = (float)src[base + i], x2 = (float)src[base + i + 64];
  dst[base + i]      = (_Float16)(x1 * c - x2 * s);
  dst[base + i + 64] = (_Float16)(x2 * c + x1 * s);
}

// ---------------------------------------------------------------------------
// Flash attention: one wave per (b, h, 16-query tile), 32-key tiles.
// K/V tiles staged into LDS by TDM (double buffered across key tiles);
// P transposed through LDS; GQA hk = h/4.
// ---------------------------------------------------------------------------
__global__ void __launch_bounds__(32)
flash_attn(const _Float16* __restrict__ Qh, const _Float16* __restrict__ Kh,
           const _Float16* __restrict__ Vh, _Float16* __restrict__ Oh) {
  __shared__ __align__(128) _Float16 smemK[2][32 * 128];
  __shared__ __align__(128) _Float16 smemV[2][32 * 128];
  __shared__ __align__(128) _Float16 smemP[16 * 32];
  const int lane  = threadIdx.x & 31;
  const int half_ = lane >> 4;
  const int lm    = lane & 15;
  const int q0 = blockIdx.x * 16;
  const int h  = blockIdx.y;
  const int b  = blockIdx.z;
  const int hk = h / (N_HEADS / N_KV);
  const int kvstride = N_KV * HEAD_DIM;   // 512 elements between tokens
  const float scale = 0.08838834764831845f;  // 1/sqrt(128)
  const float NEG_INF = -__builtin_inff();
  const _Float16* kg0 = Kh + (((size_t)(b * SEQ)) * N_KV + hk) * HEAD_DIM;
  const _Float16* vg0 = Vh + (((size_t)(b * SEQ)) * N_KV + hk) * HEAD_DIM;
  uint32_t ldsK[2] = {lds_offset_of(&smemK[0][0]), lds_offset_of(&smemK[1][0])};
  uint32_t ldsV[2] = {lds_offset_of(&smemV[0][0]), lds_offset_of(&smemV[1][0])};

  // Q tile in WMMA A layout (4 chunks of K=32); contiguous b128 loads
  v16h qa[4];
  {
    const _Float16* qrow =
        Qh + (((size_t)(b * SEQ + q0 + lm)) * N_HEADS + h) * HEAD_DIM;
#pragma unroll
    for (int c = 0; c < 4; ++c) {
#pragma unroll
      for (int e = 0; e < 8; ++e) {
        qa[c][e]     = qrow[c * 32 + half_ * 16 + e];
        qa[c][e + 8] = qrow[c * 32 + half_ * 16 + 8 + e];
      }
    }
  }

  v8f o[8];
  v8f zacc = {};
#pragma unroll
  for (int j = 0; j < 8; ++j) o[j] = zacc;
  float Mrun[8], Lrun[8];
#pragma unroll
  for (int r = 0; r < 8; ++r) { Mrun[r] = NEG_INF; Lrun[r] = 0.0f; }

  const int ktiles = (q0 + 15) / 32 + 1;
  // prologue: DMA K/V tile 0 into buffer 0
  tdm_load_2d(ldsK[0], kg0, 128u, 32u, 128u, 32u, (uint32_t)kvstride);
  tdm_load_2d(ldsV[0], vg0, 128u, 32u, 128u, 32u, (uint32_t)kvstride);

  for (int kt = 0; kt < ktiles; ++kt) {
    const int kbase = kt * 32;
    const int cur = kt & 1;
    if (kt + 1 < ktiles) {       // prefetch next K/V pair into other buffer
      const size_t nb = (size_t)(kbase + 32) * kvstride;
      tdm_load_2d(ldsK[cur ^ 1], kg0 + nb, 128u, 32u, 128u, 32u,
                  (uint32_t)kvstride);
      tdm_load_2d(ldsV[cur ^ 1], vg0 + nb, 128u, 32u, 128u, 32u,
                  (uint32_t)kvstride);
      __builtin_amdgcn_s_wait_tensorcnt(2);   // pair kt done, kt+1 in flight
    } else {
      __builtin_amdgcn_s_wait_tensorcnt(0);
    }
    __syncthreads();
    const _Float16* sk = &smemK[cur][0];
    const _Float16* sv = &smemV[cur][0];

    // ---- S = Q * K^T : two 16-key halves, 4 K-chunks each ---------------
    v8f s0 = zacc, s1 = zacc;
#pragma unroll
    for (int c = 0; c < 4; ++c) {
      v16h kb0, kb1;
#pragma unroll
      for (int e = 0; e < 16; ++e) {
        kb0[e] = sk[lm * 128 + c * 32 + half_ * 16 + e];
        kb1[e] = sk[(16 + lm) * 128 + c * 32 + half_ * 16 + e];
      }
      s0 = wmma_f16(qa[c], kb0, s0);
      s1 = wmma_f16(qa[c], kb1, s1);
    }

    // ---- online softmax (row r+8*half_ on each 16-lane group) -----------
#pragma unroll
    for (int r = 0; r < 8; ++r) {
      const int m = r + half_ * 8;
      float v0 = s0[r] * scale, v1 = s1[r] * scale;
      if (kbase + lm > q0 + m)      v0 = NEG_INF;
      if (kbase + 16 + lm > q0 + m) v1 = NEG_INF;
      float rmax = fmaxf(v0, v1);
#pragma unroll
      for (int d = 1; d < 16; d <<= 1) rmax = fmaxf(rmax, __shfl_xor(rmax, d));
      float newM  = fmaxf(Mrun[r], rmax);
      float alpha = __expf(Mrun[r] - newM);
      float p0 = __expf(v0 - newM);
      float p1 = __expf(v1 - newM);
      float rsum = p0 + p1;
#pragma unroll
      for (int d = 1; d < 16; d <<= 1) rsum += __shfl_xor(rsum, d);
      Lrun[r] = Lrun[r] * alpha + rsum;
      Mrun[r] = newM;
#pragma unroll
      for (int j = 0; j < 8; ++j) o[j][r] *= alpha;
      smemP[m * 32 + lm]      = (_Float16)p0;
      smemP[m * 32 + 16 + lm] = (_Float16)p1;
    }
    __syncthreads();

    // ---- P back from LDS in A layout (16x32) ----------------------------
    v16h pa;
#pragma unroll
    for (int e = 0; e < 8; ++e) {
      pa[e]     = smemP[lm * 32 + half_ * 16 + e];
      pa[e + 8] = smemP[lm * 32 + half_ * 16 + 8 + e];
    }
    __syncthreads();

    // ---- O += P * V : 8 dim-tiles of 16, V from LDS ---------------------
#pragma unroll
    for (int j = 0; j < 8; ++j) {
      v16h vb;
#pragma unroll
      for (int e = 0; e < 16; ++e)
        vb[e] = sv[(half_ * 16 + e) * 128 + j * 16 + lm];
      o[j] = wmma_f16(pa, vb, o[j]);
    }
  }

  // ---- normalize, store O (f16) ------------------------------------------
#pragma unroll
  for (int r = 0; r < 8; ++r) {
    float inv = 1.0f / Lrun[r];
    const int t = q0 + r + half_ * 8;
    _Float16* orow = Oh + (size_t)(b * SEQ + t) * D_MODEL + h * HEAD_DIM;
#pragma unroll
    for (int j = 0; j < 8; ++j)
      orow[j * 16 + lm] = (_Float16)(o[j][r] * inv);
  }
}

// ---------------------------------------------------------------------------
extern "C" void kernel_launch(void* const* d_in, const int* in_sizes, int n_in,
                              void* d_out, int out_size, void* d_ws,
                              size_t ws_size, hipStream_t stream) {
  const float* x  = (const float*)d_in[0];
  // d_in[1] = causal mask (bool) -- handled analytically, unused
  const float* Wq = (const float*)d_in[2];
  const float* Wk = (const float*)d_in[3];
  const float* Wv = (const float*)d_in[4];
  const float* Wo = (const float*)d_in[5];

  const int NKV_D = N_KV * HEAD_DIM;  // 512
  char* ws = (char*)d_ws;
  _Float16* xh  = (_Float16*)ws; ws += (size_t)NTOK * D_MODEL * 2;     // 16 MB
  _Float16* Wqh = (_Float16*)ws; ws += (size_t)D_MODEL * D_MODEL * 2;  //  8 MB
  _Float16* Wkh = (_Float16*)ws; ws += (size_t)D_MODEL * NKV_D * 2;    //  2 MB
  _Float16* Wvh = (_Float16*)ws; ws += (size_t)D_MODEL * NKV_D * 2;    //  2 MB
  _Float16* Woh = (_Float16*)ws; ws += (size_t)D_MODEL * D_MODEL * 2;  //  8 MB
  _Float16* Qp  = (_Float16*)ws; ws += (size_t)NTOK * D_MODEL * 2;     // 16 MB
  _Float16* Kp  = (_Float16*)ws; ws += (size_t)NTOK * NKV_D * 2;       //  4 MB
  _Float16* Qh  = (_Float16*)ws; ws += (size_t)NTOK * D_MODEL * 2;     // 16 MB
  _Float16* Kh  = (_Float16*)ws; ws += (size_t)NTOK * NKV_D * 2;       //  4 MB
  _Float16* Vh  = (_Float16*)ws; ws += (size_t)NTOK * NKV_D * 2;       //  4 MB
  _Float16* Oh  = (_Float16*)ws; ws += (size_t)NTOK * D_MODEL * 2;     // 16 MB

  const dim3 wave(32);

  // one-time f16 reformat of activations and weights
  int nx = NTOK * D_MODEL;
  cast_h<<<(nx + 255) / 256, 256, 0, stream>>>(x, xh, nx);
  int nwq = D_MODEL * D_MODEL;
  cast_h<<<(nwq + 255) / 256, 256, 0, stream>>>(Wq, Wqh, nwq);
  int nwk = D_MODEL * NKV_D;
  cast_h<<<(nwk + 255) / 256, 256, 0, stream>>>(Wk, Wkh, nwk);
  cast_h<<<(nwk + 255) / 256, 256, 0, stream>>>(Wv, Wvh, nwk);
  cast_h<<<(nwq + 255) / 256, 256, 0, stream>>>(Wo, Woh, nwq);

  // projections (f16 WMMA, TDM-staged B slabs, f16 out)
  gemm_h16<_Float16><<<dim3(D_MODEL / 64, NTOK / 16), wave, 0, stream>>>(
      xh, Wqh, Qp, NTOK, D_MODEL, D_MODEL);
  gemm_h16<_Float16><<<dim3(NKV_D / 64, NTOK / 16), wave, 0, stream>>>(
      xh, Wkh, Kp, NTOK, NKV_D, D_MODEL);
  gemm_h16<_Float16><<<dim3(NKV_D / 64, NTOK / 16), wave, 0, stream>>>(
      xh, Wvh, Vh, NTOK, NKV_D, D_MODEL);

  // RoPE on Q and K
  int tq = NTOK * N_HEADS * 64;
  rope_h<<<(tq + 255) / 256, 256, 0, stream>>>(Qp, Qh, N_HEADS, tq);
  int tk = NTOK * N_KV * 64;
  rope_h<<<(tk + 255) / 256, 256, 0, stream>>>(Kp, Kh, N_KV, tk);

  // flash attention (causal, GQA, double-buffered TDM K/V)
  flash_attn<<<dim3(SEQ / 16, N_HEADS, BATCH), wave, 0, stream>>>(Qh, Kh, Vh, Oh);

  // output projection -> d_out (fp32)
  gemm_h16<float><<<dim3(D_MODEL / 64, NTOK / 16), wave, 0, stream>>>(
      Oh, Woh, (float*)d_out, NTOK, D_MODEL, D_MODEL);
}
